// Decoder_41437844472423
// MI455X (gfx1250) — matrix operand order repaired and checked by reading, not compiled, not run on hardware
//
#include <hip/hip_runtime.h>
#include <hip/hip_bf16.h>

// ---------------- WMMA types ----------------
typedef __attribute__((ext_vector_type(16))) __bf16 v16bf;
typedef __attribute__((ext_vector_type(8)))  float  v8f;

// ---------------- problem constants ----------------
#define BATCH 8
#define TIN 200
#define NSTEPS 200
#define IN_FEAT 512
#define CTXD 544          // 512 + 32
#define QDIM 1024
#define DECD 512
#define PREND 256
#define ATT_K 2016        // 992 (x) + 1024 (h)
#define ATT_N 4096
#define DEC_K 2272        // 1760 (m2) + 512 (dh)
#define DEC_N 2048
#define ATT_KT 63         // 2016/32
#define DEC_KT 71         // 2272/32
#define ATT_NT 256        // 4096/16
#define DEC_NT 128        // 2048/16
#define PROJ_K 1248
#define PROJ_N 160
#define TA_K 1760

#define NWG 32
#define TPB 256
#define NTHREADS (NWG*TPB)

// ---------------- ws layout (bytes, all 256-aligned) ----------------
#define OFF_BAR    ((size_t)0)
#define OFF_QUERY  ((size_t)256)                         // 8*1024 f32
#define OFF_QC     (OFF_QUERY + 32768)
#define OFF_DH     (OFF_QC    + 32768)                   // 8*512
#define OFF_DC     (OFF_DH    + 16384)
#define OFF_CTX    (OFF_DC    + 16384)                   // 8*544
#define OFF_ALPHA  (OFF_CTX   + 17408)                   // 8*200
#define OFF_U      (OFF_ALPHA + 6400)
#define OFF_SQ     (OFF_U     + 256)
#define OFF_GA     (OFF_SQ    + 256)                     // 8*4096
#define OFF_GD     (OFF_GA    + 131072)                  // 8*2048
#define OFF_BIASA  (OFF_GD    + 65536)                   // 4096
#define OFF_BIASD  (OFF_BIASA + 16384)                   // 2048
#define OFF_XATT   (OFF_BIASD + 8192)                    // 63 KB tiles (bf16)
#define OFF_XDEC   (OFF_XATT  + (size_t)ATT_KT*1024)     // 71 KB tiles
#define OFF_PRE    (OFF_XDEC  + (size_t)DEC_KT*1024)     // 200*8*256 f32
#define OFF_HT     (OFF_PRE   + (size_t)NSTEPS*BATCH*PREND*4)
#define OFF_PORD   (OFF_HT    + (size_t)NSTEPS*BATCH*PREND*4)  // 8*200*3*32 f32
#define OFF_WATT   (OFF_PORD  + (size_t)BATCH*TIN*3*32*4)      // 256*63 KB bf16 tiles
#define OFF_WDEC   (OFF_WATT  + (size_t)ATT_NT*ATT_KT*1024)    // 128*71 KB

// ---------------- helpers ----------------
__device__ __forceinline__ unsigned short f2bf(float f) {
    unsigned u = __float_as_uint(f);
    unsigned r = (u + 0x7FFFu + ((u >> 16) & 1u)) >> 16;
    return (unsigned short)r;
}
__device__ __forceinline__ float sigm(float x) { return 1.0f / (1.0f + __expf(-x)); }

// byte offset of A-matrix element (m, k_local) inside a 1KB 16x32 bf16 WMMA A tile
__device__ __forceinline__ int a_tile_off(int m, int kl) {
    int lane = m + (kl >> 4) * 16;
    return lane * 32 + (kl & 15) * 2;
}

__device__ __forceinline__ float wave_reduce(float v) {
    #pragma unroll
    for (int off = 16; off > 0; off >>= 1) v += __shfl_down(v, off);
    return v;
}

// ---------------- grid barrier (sense via generation counter) ----------------
__device__ void gbar(unsigned* bar) {
    __syncthreads();
    if (threadIdx.x == 0) {
        __threadfence();
        unsigned gen = ((volatile unsigned*)bar)[1];
        if (atomicAdd(&bar[0], 1u) == (unsigned)(NWG - 1)) {
            atomicExch(&bar[0], 0u);
            __threadfence();
            atomicAdd(&bar[1], 1u);
        } else {
            while (((volatile unsigned*)bar)[1] == gen) { __builtin_amdgcn_s_sleep(8); }
        }
        __threadfence();
    }
    __syncthreads();
}

// ================= prep kernels =================
__global__ __launch_bounds__(TPB) void k_init(char* ws,
        const float* att_init, const float* dec_init,
        const float* att_bih, const float* att_bhh,
        const float* dec_bih, const float* dec_bhh) {
    int tid = blockIdx.x * blockDim.x + threadIdx.x;
    if (tid < 16) ((unsigned*)(ws + OFF_BAR))[tid] = 0u;
    if (tid < 8192) { // query / qc
        int b = tid / QDIM, j = tid % QDIM;
        ((float*)(ws + OFF_QUERY))[b * QDIM + j] = att_init[j];
        ((float*)(ws + OFF_QC))[b * QDIM + j] = 0.0f;
    }
    if (tid < 4096) { // dh / dc
        int b = tid / DECD, j = tid % DECD;
        ((float*)(ws + OFF_DH))[b * DECD + j] = dec_init[j];
        ((float*)(ws + OFF_DC))[b * DECD + j] = 0.0f;
    }
    if (tid < BATCH * CTXD) ((float*)(ws + OFF_CTX))[tid] = 0.0f;
    if (tid < BATCH * TIN) {
        int i = tid % TIN;
        ((float*)(ws + OFF_ALPHA))[tid] = (i == 0) ? 1.0f : 1e-7f;
    }
    if (tid < BATCH) {
        ((float*)(ws + OFF_U))[tid]  = 0.5f;
        ((float*)(ws + OFF_SQ))[tid] = 1.4f;
    }
    if (tid < ATT_N) ((float*)(ws + OFF_BIASA))[tid] = att_bih[tid] + att_bhh[tid];
    if (tid < DEC_N) ((float*)(ws + OFF_BIASD))[tid] = dec_bih[tid] + dec_bhh[tid];
}

// pack att+dec LSTM weights (fp32 row-major) into bf16 WMMA B-tiles.
// tile element e = lane*16+idx  maps to  n_local = lane%16, k_local = (lane/16)*16 + idx
__global__ __launch_bounds__(TPB) void k_pack(char* ws,
        const float* aWih, const float* aWhh, const float* dWih, const float* dWhh) {
    const long attE = (long)ATT_NT * ATT_KT * 512;
    const long decE = (long)DEC_NT * DEC_KT * 512;
    long tid = (long)blockIdx.x * blockDim.x + threadIdx.x;
    if (tid >= attE + decE) return;
    if (tid < attE) {
        long tile = tid / 512; int e = (int)(tid % 512);
        int nT = (int)(tile / ATT_KT), kT = (int)(tile % ATT_KT);
        int lane = e / 16, idx = e % 16;
        int n = nT * 16 + (lane % 16);
        int k = kT * 32 + (lane / 16) * 16 + idx;
        float w = (k < 992) ? aWih[(long)n * 992 + k] : aWhh[(long)n * 1024 + (k - 992)];
        ((unsigned short*)(ws + OFF_WATT))[tid] = f2bf(w);
    } else {
        long t2 = tid - attE;
        long tile = t2 / 512; int e = (int)(t2 % 512);
        int nT = (int)(tile / DEC_KT), kT = (int)(tile % DEC_KT);
        int lane = e / 16, idx = e % 16;
        int n = nT * 16 + (lane % 16);
        int k = kT * 32 + (lane / 16) * 16 + idx;
        float w = (k < 1760) ? dWih[(long)n * 1760 + k] : dWhh[(long)n * 512 + (k - 1760)];
        ((unsigned short*)(ws + OFF_WDEC))[t2] = f2bf(w);
    }
}

// prenet stage 1: h = relu(ms @ W1^T), ms row (t,b): t==0 -> mem_init else memory step
__global__ __launch_bounds__(TPB) void k_prenet1(char* ws, const float* memory,
        const float* mem_init, const float* W1) {
    int tid = blockIdx.x * blockDim.x + threadIdx.x;
    if (tid >= NSTEPS * BATCH * PREND) return;
    int j = tid % PREND, row = tid / PREND;
    int b = row % BATCH, t = row / BATCH;
    const float* ms = (t == 0) ? mem_init : (memory + (long)b * 32000 + (long)(t - 1) * 160);
    float acc = 0.0f;
    for (int q = 0; q < 160; ++q) acc += ms[q] * W1[j * 160 + q];
    ((float*)(ws + OFF_HT))[tid] = fmaxf(acc, 0.0f);
}
// prenet stage 2: pre = relu(h @ W2^T)
__global__ __launch_bounds__(TPB) void k_prenet2(char* ws, const float* W2) {
    int tid = blockIdx.x * blockDim.x + threadIdx.x;
    if (tid >= NSTEPS * BATCH * PREND) return;
    int j = tid % PREND, row = tid / PREND;
    const float* h = (const float*)(ws + OFF_HT) + (long)row * PREND;
    float acc = 0.0f;
    for (int q = 0; q < PREND; ++q) acc += h[q] * W2[j * PREND + q];
    ((float*)(ws + OFF_PRE))[tid] = fmaxf(acc, 0.0f);
}

// P[b,t,k,o] = sum_i ord_W[o,i,k] * inputs[b,t,i]   (step-invariant conv factor)
__global__ __launch_bounds__(TPB) void k_pord(char* ws, const float* inputs, const float* ordW) {
    int tid = blockIdx.x * blockDim.x + threadIdx.x;
    if (tid >= BATCH * TIN * 3 * 32) return;
    int o = tid % 32, k = (tid / 32) % 3, t = (tid / 96) % TIN, b = tid / (96 * TIN);
    const float* x = inputs + ((long)b * TIN + t) * IN_FEAT;
    float acc = 0.0f;
    for (int i = 0; i < IN_FEAT; ++i) acc += ordW[(long)o * 1536 + i * 3 + k] * x[i];
    ((float*)(ws + OFF_PORD))[tid] = acc;
}

// ================= persistent decoder =================
__global__ __launch_bounds__(TPB) void k_decoder(char* ws, float* dout,
        const float* inputs, const float* style, const float* speaker,
        const float* ordb, const float* bng, const float* bnb,
        const float* tauW, const float* taub, const float* tasW, const float* tasb,
        const float* projW, const float* projb) {
    unsigned* bar = (unsigned*)(ws + OFF_BAR);
    const int gtid = blockIdx.x * blockDim.x + threadIdx.x;
    const int gwave = gtid >> 5;
    const int lane = threadIdx.x & 31;

    float* query = (float*)(ws + OFF_QUERY);
    float* qc    = (float*)(ws + OFF_QC);
    float* dh    = (float*)(ws + OFF_DH);
    float* dc    = (float*)(ws + OFF_DC);
    float* ctx   = (float*)(ws + OFF_CTX);
    float* alpha = (float*)(ws + OFF_ALPHA);
    float* uP    = (float*)(ws + OFF_U);
    float* sqP   = (float*)(ws + OFF_SQ);
    float* GA    = (float*)(ws + OFF_GA);
    float* GD    = (float*)(ws + OFF_GD);
    float* biasA = (float*)(ws + OFF_BIASA);
    float* biasD = (float*)(ws + OFF_BIASD);
    float* Pord  = (float*)(ws + OFF_PORD);
    float* pre   = (float*)(ws + OFF_PRE);
    float* aligns = dout + 256000;

    const float bscale0 = 1.0f / sqrtf(1.0f + 1e-5f);

    for (int t = 0; t < NSTEPS; ++t) {
        // -------- P0: alpha update (waves 0..7) + build att A-tiles (WG1..31) ----
        if (gwave < BATCH) {
            int b = gwave;
            float u = uP[b], sq = sqP[b];
            float av[7]; float psum = 0.0f;
            #pragma unroll
            for (int c = 0; c < 7; ++c) {
                int i = lane + c * 32;
                float a = 0.0f;
                if (i < TIN) {
                    float ap = (i == 0) ? 0.0f : alpha[b * TIN + i - 1];
                    float base = (1.0f - u) * alpha[b * TIN + i] + u * ap + 1e-6f;
                    a = exp2f(sq * log2f(base));
                }
                av[c] = a; psum += a;
            }
            float s = wave_reduce(psum);
            s = __shfl(s, 0);
            float inv = 1.0f / s;
            #pragma unroll
            for (int c = 0; c < 7; ++c) {
                int i = lane + c * 32;
                if (i < TIN) {
                    float a = av[c] * inv;
                    alpha[b * TIN + i] = a;
                    aligns[(long)b * 40000 + (long)t * TIN + i] = a;
                }
            }
        }
        if (gtid >= TPB) {
            unsigned short* xa = (unsigned short*)(ws + OFF_XATT);
            for (int e = gtid - TPB; e < 16 * ATT_K; e += NTHREADS - TPB) {
                int m = e / ATT_K, k = e % ATT_K;
                float v = 0.0f;
                if (m < BATCH) {
                    if (k < 256)       v = pre[((long)t * BATCH + m) * PREND + k];
                    else if (k < 800)  v = ctx[m * CTXD + k - 256];
                    else if (k < 928)  v = style[m * 128 + k - 800];
                    else if (k < 992)  v = speaker[m * 64 + k - 928];
                    else               v = query[m * QDIM + k - 992];
                }
                int kT = k / 32;
                *(unsigned short*)((char*)xa + kT * 1024 + a_tile_off(m, k % 32)) = f2bf(v);
            }
        }
        gbar(bar);

        // -------- P1: att gates GEMM (256 waves) + ctx + order ----
        {
            const char* wb = ws + OFF_WATT + (size_t)gwave * ATT_KT * 1024;
            const char* ab = ws + OFF_XATT;
            v8f acc = {};
            for (int kT = 0; kT < ATT_KT; ++kT) {
                v16bf a = *(const v16bf*)(ab + kT * 1024 + lane * 32);
                v16bf b = *(const v16bf*)(wb + kT * 1024 + lane * 32);
                __builtin_prefetch(wb + (kT + 1) * 1024 + lane * 32, 0, 1);
                acc = __builtin_amdgcn_wmma_f32_16x16x32_bf16(false, a, false, b,
                                                              (short)0, acc, false, false);
            }
            if (lane < 16) {
                int n = gwave * 16 + lane;
                float bs = biasA[n];
                #pragma unroll
                for (int r = 0; r < 8; ++r) GA[r * ATT_N + n] = acc[r] + bs;
            }
        }
        if (gtid < BATCH * IN_FEAT) {           // ctx = alpha @ inputs
            int b = gtid / IN_FEAT, i = gtid % IN_FEAT;
            float acc = 0.0f;
            const float* xin = inputs + (long)b * TIN * IN_FEAT + i;
            const float* al = alpha + b * TIN;
            for (int tt = 0; tt < TIN; ++tt) acc += al[tt] * xin[(long)tt * IN_FEAT];
            ctx[b * CTXD + i] = acc;
        } else if (gtid < BATCH * IN_FEAT + BATCH * 32) {   // order net (factored)
            int idx = gtid - BATCH * IN_FEAT;
            int b = idx / 32, o = idx % 32;
            float sc = bng[o] * bscale0, bb = bnb[o], ob = ordb[o];
            const float* al = alpha + b * TIN;
            float tot = 0.0f;
            for (int tt = 0; tt < TIN; ++tt) {
                float s = 0.0f;
                #pragma unroll
                for (int k = 0; k < 3; ++k) {
                    int p = tt - 1 + k;
                    if (p >= 0 && p < TIN)
                        s += al[p] * Pord[(((long)b * TIN + p) * 3 + k) * 32 + o];
                }
                float y = (s + ob) * sc + bb;
                tot += fmaxf(y, 0.0f);
            }
            ctx[b * CTXD + IN_FEAT + o] = tot;
        }
        gbar(bar);

        // -------- P2: att LSTM pointwise ----
        {
            int b = gtid / QDIM, j = gtid % QDIM;
            float gi = GA[b * ATT_N + j];
            float gf = GA[b * ATT_N + QDIM + j];
            float gg = GA[b * ATT_N + 2 * QDIM + j];
            float go = GA[b * ATT_N + 3 * QDIM + j];
            float c2 = sigm(gf) * qc[b * QDIM + j] + sigm(gi) * tanhf(gg);
            query[b * QDIM + j] = sigm(go) * tanhf(c2);
            qc[b * QDIM + j] = c2;
        }
        gbar(bar);

        // -------- P3: TA dots (waves 0..15) + build dec A-tiles ----
        if (gwave < 16) {
            int b = gwave >> 1, which = gwave & 1;
            const float* W = which ? tasW : tauW;
            float acc = 0.0f;
            for (int k = lane; k < TA_K; k += 32) {
                float x;
                if (k < 544)        x = ctx[b * CTXD + k];
                else if (k < 1568)  x = query[b * QDIM + k - 544];
                else if (k < 1696)  x = style[b * 128 + k - 1568];
                else                x = speaker[b * 64 + k - 1696];
                acc += x * W[k];
            }
            acc = wave_reduce(acc);
            if (lane == 0) {
                if (which) sqP[b] = sigm(acc + tasb[0]) + 1.0f;
                else       uP[b]  = sigm(acc + taub[0]);
            }
        }
        if (gtid >= 2 * TPB) {
            unsigned short* xd = (unsigned short*)(ws + OFF_XDEC);
            for (int e = gtid - 2 * TPB; e < 16 * DEC_K; e += NTHREADS - 2 * TPB) {
                int m = e / DEC_K, k = e % DEC_K;
                float v = 0.0f;
                if (m < BATCH) {
                    if (k < 1024)      v = query[m * QDIM + k];
                    else if (k < 1568) v = ctx[m * CTXD + k - 1024];
                    else if (k < 1696) v = style[m * 128 + k - 1568];
                    else if (k < 1760) v = speaker[m * 64 + k - 1696];
                    else               v = dh[m * DECD + k - 1760];
                }
                int kT = k / 32;
                *(unsigned short*)((char*)xd + kT * 1024 + a_tile_off(m, k % 32)) = f2bf(v);
            }
        }
        gbar(bar);

        // -------- P4: dec gates GEMM (128 waves) ----
        if (gwave < DEC_NT) {
            const char* wb = ws + OFF_WDEC + (size_t)gwave * DEC_KT * 1024;
            const char* ab = ws + OFF_XDEC;
            v8f acc = {};
            for (int kT = 0; kT < DEC_KT; ++kT) {
                v16bf a = *(const v16bf*)(ab + kT * 1024 + lane * 32);
                v16bf b = *(const v16bf*)(wb + kT * 1024 + lane * 32);
                __builtin_prefetch(wb + (kT + 1) * 1024 + lane * 32, 0, 1);
                acc = __builtin_amdgcn_wmma_f32_16x16x32_bf16(false, a, false, b,
                                                              (short)0, acc, false, false);
            }
            if (lane < 16) {
                int n = gwave * 16 + lane;
                float bs = biasD[n];
                #pragma unroll
                for (int r = 0; r < 8; ++r) GD[r * DEC_N + n] = acc[r] + bs;
            }
        }
        gbar(bar);

        // -------- P5: dec LSTM pointwise ----
        if (gtid < BATCH * DECD) {
            int b = gtid / DECD, j = gtid % DECD;
            float gi = GD[b * DEC_N + j];
            float gf = GD[b * DEC_N + DECD + j];
            float gg = GD[b * DEC_N + 2 * DECD + j];
            float go = GD[b * DEC_N + 3 * DECD + j];
            float c2 = sigm(gf) * dc[b * DECD + j] + sigm(gi) * tanhf(gg);
            dh[b * DECD + j] = sigm(go) * tanhf(c2);
            dc[b * DECD + j] = c2;
        }
        gbar(bar);

        // -------- P6: projection + scatter to output layout (8,80,400) ----
        if (gtid < BATCH * PROJ_N) {
            int b = gtid / PROJ_N, p = gtid % PROJ_N;
            const float* W = projW + (long)p * PROJ_K;
            float acc = projb[p];
            for (int k = 0; k < PROJ_K; ++k) {
                float x;
                if (k < 512)        x = dh[b * DECD + k];
                else if (k < 1056)  x = ctx[b * CTXD + k - 512];
                else if (k < 1184)  x = style[b * 128 + k - 1056];
                else                x = speaker[b * 64 + k - 1184];
                acc += x * W[k];
            }
            int j = p / 80, c = p % 80;
            dout[(long)b * 32000 + (long)c * 400 + 2 * t + j] = acc;
        }
        gbar(bar);
    }
}

// ================= launch =================
extern "C" void kernel_launch(void* const* d_in, const int* in_sizes, int n_in,
                              void* d_out, int out_size, void* d_ws, size_t ws_size,
                              hipStream_t stream) {
    const float* inputs  = (const float*)d_in[0];
    const float* memory  = (const float*)d_in[1];
    const float* style   = (const float*)d_in[2];
    const float* speaker = (const float*)d_in[3];
    // d_in[4] = mask (all ones, unused)
    const float* prenet_W1 = (const float*)d_in[5];
    const float* prenet_W2 = (const float*)d_in[6];
    const float* att_Wih = (const float*)d_in[7];
    const float* att_Whh = (const float*)d_in[8];
    const float* att_bih = (const float*)d_in[9];
    const float* att_bhh = (const float*)d_in[10];
    const float* ord_W   = (const float*)d_in[11];
    const float* ord_b   = (const float*)d_in[12];
    const float* bn_g    = (const float*)d_in[13];
    const float* bn_b    = (const float*)d_in[14];
    const float* ta_u_W  = (const float*)d_in[15];
    const float* ta_u_b  = (const float*)d_in[16];
    const float* ta_sq_W = (const float*)d_in[17];
    const float* ta_sq_b = (const float*)d_in[18];
    const float* dec_Wih = (const float*)d_in[19];
    const float* dec_Whh = (const float*)d_in[20];
    const float* dec_bih = (const float*)d_in[21];
    const float* dec_bhh = (const float*)d_in[22];
    const float* proj_W  = (const float*)d_in[23];
    const float* proj_b  = (const float*)d_in[24];
    const float* att_init = (const float*)d_in[25];
    const float* mem_init = (const float*)d_in[26];
    const float* dec_init = (const float*)d_in[27];

    char* ws = (char*)d_ws;
    float* out = (float*)d_out;

    hipLaunchKernelGGL(k_init, dim3(NWG), dim3(TPB), 0, stream,
                       ws, att_init, dec_init, att_bih, att_bhh, dec_bih, dec_bhh);

    long packE = (long)ATT_NT * ATT_KT * 512 + (long)DEC_NT * DEC_KT * 512;
    hipLaunchKernelGGL(k_pack, dim3((unsigned)((packE + TPB - 1) / TPB)), dim3(TPB), 0, stream,
                       ws, att_Wih, att_Whh, dec_Wih, dec_Whh);

    int preE = NSTEPS * BATCH * PREND;
    hipLaunchKernelGGL(k_prenet1, dim3((preE + TPB - 1) / TPB), dim3(TPB), 0, stream,
                       ws, memory, mem_init, prenet_W1);
    hipLaunchKernelGGL(k_prenet2, dim3((preE + TPB - 1) / TPB), dim3(TPB), 0, stream,
                       ws, prenet_W2);

    int pE = BATCH * TIN * 3 * 32;
    hipLaunchKernelGGL(k_pord, dim3((pE + TPB - 1) / TPB), dim3(TPB), 0, stream,
                       ws, inputs, ord_W);

    hipLaunchKernelGGL(k_decoder, dim3(NWG), dim3(TPB), 0, stream,
                       ws, out, inputs, style, speaker,
                       ord_b, bn_g, bn_b,
                       ta_u_W, ta_u_b, ta_sq_W, ta_sq_b,
                       proj_W, proj_b);
}
